// SamClip_2740189135632
// MI455X (gfx1250) — 8-bit matrix operand order repaired and checked by reading, compiled, not run on hardware
//
#include <hip/hip_runtime.h>
#include <hip/hip_bf16.h>
#include <stdint.h>

typedef __attribute__((ext_vector_type(16))) int   v16i;
typedef __attribute__((ext_vector_type(8)))  float v8f;

#define N_MASKS 1024
#define HW      65536
#define KSPLIT  2
#define KCHUNK  (HW / KSPLIT)
#define KSTEPS  (KCHUNK / 128)
#define LDS_STRIDE 36
#define LDS_TILE   (192 * LDS_STRIDE)
#define FP8_ONE 0x38u

__global__ void rank_kernel(const float* __restrict__ scores,
                            int* __restrict__ order,
                            float* __restrict__ ssort) {
    int i = blockIdx.x * blockDim.x + threadIdx.x;
    if (i >= N_MASKS) return;
    float si = scores[i];
    int rank = 0;
    for (int j = 0; j < N_MASKS; ++j) {
        float sj = scores[j];
        rank += (sj > si) || (sj == si && j < i);
    }
    order[rank] = i;
    ssort[rank] = si;
}

__global__ void pack_kernel(const float* __restrict__ masks,
                            const int* __restrict__ order,
                            unsigned* __restrict__ P) {
    int r = blockIdx.x;
    int src = order[r];
    const float4* in = (const float4*)(masks + (size_t)src * HW);
    unsigned* out = P + (size_t)r * (HW / 4);
    for (int c = threadIdx.x; c < HW / 4; c += blockDim.x) {
        float4 m = in[c];
        unsigned b = 0;
        b |= (m.x > 0.5f) ? (FP8_ONE)       : 0u;
        b |= (m.y > 0.5f) ? (FP8_ONE << 8)  : 0u;
        b |= (m.z > 0.5f) ? (FP8_ONE << 16) : 0u;
        b |= (m.w > 0.5f) ? (FP8_ONE << 24) : 0u;
        out[c] = b;
    }
}

__launch_bounds__(256)
__global__ void gemm_kernel(const unsigned char* __restrict__ P,
                            float* __restrict__ inter_parts) {
    const int brow = blockIdx.y * 64;
    const int bcol = blockIdx.x * 128;
    if (bcol + 128 <= brow) return;

    __shared__ unsigned lds[2][LDS_TILE];

    const int tid  = threadIdx.x;
    const int lane = tid & 31;
    const int wave = tid >> 5;
    const int wr   = wave & 1;
    const int wc   = wave >> 1;
    const int l15  = lane & 15;
    const int h    = lane >> 4;
    const size_t kbase = (size_t)blockIdx.z * KCHUNK;

    const unsigned char* gA[2]; int lA[2];
    #pragma unroll
    for (int s = 0; s < 2; ++s) {
        int ch = tid + s * 256, row = ch >> 3, off = (ch & 7) * 16;
        gA[s] = P + (size_t)(brow + row) * HW + kbase + off;
        lA[s] = row * LDS_STRIDE + (off >> 2);
    }
    const unsigned char* gB[4]; int lB[4];
    #pragma unroll
    for (int s = 0; s < 4; ++s) {
        int ch = tid + s * 256, row = ch >> 3, off = (ch & 7) * 16;
        gB[s] = P + (size_t)(bcol + row) * HW + kbase + off;
        lB[s] = 64 * LDS_STRIDE + row * LDS_STRIDE + (off >> 2);
    }

    v8f acc[2][2] = {};
    uint4 rA[2], rB[4];

    #pragma unroll
    for (int s = 0; s < 2; ++s) rA[s] = *(const uint4*)gA[s];
    #pragma unroll
    for (int s = 0; s < 4; ++s) rB[s] = *(const uint4*)gB[s];
    #pragma unroll
    for (int s = 0; s < 2; ++s) *(uint4*)&lds[0][lA[s]] = rA[s];
    #pragma unroll
    for (int s = 0; s < 4; ++s) *(uint4*)&lds[0][lB[s]] = rB[s];

    int cur = 0;
    for (int k = 0; k < KSTEPS; ++k) {
        __syncthreads();
        const bool more = (k + 1 < KSTEPS);
        if (more) {
            size_t koff = (size_t)(k + 1) * 128;
            #pragma unroll
            for (int s = 0; s < 2; ++s) rA[s] = *(const uint4*)(gA[s] + koff);
            #pragma unroll
            for (int s = 0; s < 4; ++s) rB[s] = *(const uint4*)(gB[s] + koff);
        }
        const unsigned* ldsA = &lds[cur][0];
        const unsigned* ldsB = &lds[cur][64 * LDS_STRIDE];

        v16i afr[2];
        #pragma unroll
        for (int a = 0; a < 2; ++a) {
            int m = (wr * 2 + a) * 16 + l15;
            const unsigned* base = &ldsA[m * LDS_STRIDE + 2 * h];
            #pragma unroll
            for (int v = 0; v < 16; ++v) {
                int vv = v & 7;
                int kd = 16 * (v >> 3) + (vv >> 1) * 4 + (vv & 1);
                afr[a][v] = (int)base[kd];
            }
        }
        v16i bfr[2];
        #pragma unroll
        for (int b = 0; b < 2; ++b) {
            int n = (wc * 2 + b) * 16 + l15;
            const unsigned* base = &ldsB[n * LDS_STRIDE + 2 * h];
            #pragma unroll
            for (int v = 0; v < 16; ++v) {
                int vv = v & 7;
                int kd = 16 * (v >> 3) + (vv >> 1) * 4 + (vv & 1);
                bfr[b][v] = (int)base[kd];
            }
        }
        #pragma unroll
        for (int a = 0; a < 2; ++a)
            #pragma unroll
            for (int b = 0; b < 2; ++b)
                acc[a][b] = __builtin_amdgcn_wmma_f32_16x16x128_fp8_fp8(
                    afr[a], bfr[b], (short)0, acc[a][b], false, false);

        if (more) {
            int nxt = cur ^ 1;
            #pragma unroll
            for (int s = 0; s < 2; ++s) *(uint4*)&lds[nxt][lA[s]] = rA[s];
            #pragma unroll
            for (int s = 0; s < 4; ++s) *(uint4*)&lds[nxt][lB[s]] = rB[s];
        }
        cur ^= 1;
    }

    float* outP = inter_parts + (size_t)blockIdx.z * N_MASKS * N_MASKS;
    #pragma unroll
    for (int a = 0; a < 2; ++a) {
        int rowbase = brow + (wr * 2 + a) * 16 + 8 * h;
        #pragma unroll
        for (int b = 0; b < 2; ++b) {
            int col = bcol + (wc * 2 + b) * 16 + l15;
            #pragma unroll
            for (int p = 0; p < 8; ++p)
                outP[(size_t)(rowbase + p) * N_MASKS + col] = acc[a][b][p];
            float4 lo = make_float4(acc[a][b][0], acc[a][b][1], acc[a][b][2], acc[a][b][3]);
            float4 hi = make_float4(acc[a][b][4], acc[a][b][5], acc[a][b][6], acc[a][b][7]);
            float* m = outP + (size_t)col * N_MASKS + rowbase;
            *(float4*)(m)     = lo;
            *(float4*)(m + 4) = hi;
        }
    }
}

__global__ void post_kernel(const float* __restrict__ i0,
                            const float* __restrict__ i1,
                            const int* __restrict__ order,
                            const float* __restrict__ ssort,
                            float* __restrict__ out) {
    const int j = blockIdx.x;
    const int t = threadIdx.x;
    const float a_j = i0[(size_t)j * N_MASKS + j] + i1[(size_t)j * N_MASKS + j];

    float iou_max = 0.f, inn_u = 0.f, inn_l = 0.f;
    for (int i = t; i < N_MASKS; i += 256) {
        float it  = i0[(size_t)j * N_MASKS + i] + i1[(size_t)j * N_MASKS + i];
        float a_i = i0[(size_t)i * N_MASKS + i] + i1[(size_t)i * N_MASKS + i];
        float iou = it / (a_i + a_j - it + 1e-8f);
        float ri  = it / (a_i + 1e-8f);
        float rj  = it / (a_j + 1e-8f);
        float vals  = 1.0f - ri * rj;
        float inner = (ri < 0.5f && rj >= 0.85f) ? vals : 0.0f;
        if (i < j)      { iou_max = fmaxf(iou_max, iou); inn_u = fmaxf(inn_u, inner); }
        if (i >= j - 1) { inn_l = fmaxf(inn_l, inner); }
    }
    __shared__ float red[3][256];
    red[0][t] = iou_max; red[1][t] = inn_u; red[2][t] = inn_l;
    __syncthreads();
    for (int s = 128; s > 0; s >>= 1) {
        if (t < s) {
            red[0][t] = fmaxf(red[0][t], red[0][t + s]);
            red[1][t] = fmaxf(red[1][t], red[1][t + s]);
            red[2][t] = fmaxf(red[2][t], red[2][t + s]);
        }
        __syncthreads();
    }
    if (t == 0) {
        bool anyconf = ssort[0] > 0.7f;
        bool kc = anyconf ? (ssort[j] > 0.7f) : (j < 3);
        bool keep = (red[0][0] <= 0.8f) && kc &&
                    (red[1][0] <= 0.5f) && (red[2][0] <= 0.5f);
        out[j]           = (float)order[j];
        out[N_MASKS + j] = keep ? 1.0f : 0.0f;
    }
}

extern "C" void kernel_launch(void* const* d_in, const int* in_sizes, int n_in,
                              void* d_out, int out_size, void* d_ws, size_t ws_size,
                              hipStream_t stream) {
    const float* masks  = (const float*)d_in[0];
    const float* scores = (const float*)d_in[1];
    float* out = (float*)d_out;

    char* ws = (char*)d_ws;
    int*            order = (int*)ws;
    float*          ssort = (float*)(ws + 4096);
    unsigned char*  P     = (unsigned char*)(ws + 8192);
    float*          inter = (float*)(ws + 8192 + (size_t)N_MASKS * HW);
    float*          i0    = inter;
    float*          i1    = inter + (size_t)N_MASKS * N_MASKS;

    rank_kernel<<<4, 256, 0, stream>>>(scores, order, ssort);
    pack_kernel<<<N_MASKS, 256, 0, stream>>>(masks, order, (unsigned*)P);
    gemm_kernel<<<dim3(N_MASKS / 128, N_MASKS / 64, KSPLIT), 256, 0, stream>>>(P, inter);
    post_kernel<<<N_MASKS, 256, 0, stream>>>(i0, i1, order, ssort, out);
}
